// I2GNN_56616258896126
// MI455X (gfx1250) — matrix operand order repaired and checked by reading, not compiled, not run on hardware
//
#include <hip/hip_runtime.h>
#include <cstddef>

// ---------------- problem constants ----------------
#define N_NODES 15000
#define N_EDGES 60000
#define N_SG2   3000
#define N_SG    600
#define N_GRAPHS 64
#define KPAD  160   // Aact row: 128 MLP feats + 1 bias col + pad (5 K-chunks of 32)
#define WROW  168   // W2T row: 160 used + pad to 336B (84 dwords -> conflict-optimal LDS stride)
#define HCH   16    // h[src] LDS chunk rows

typedef __attribute__((ext_vector_type(8)))  float        v8f;
typedef __attribute__((ext_vector_type(16))) __bf16       bf16x16;
typedef __attribute__((ext_vector_type(8)))  __bf16       bf16x8;
typedef __attribute__((ext_vector_type(4)))  unsigned int u32x4;
typedef __attribute__((ext_vector_type(8)))  int          i32x8;
typedef __attribute__((ext_vector_type(4)))  int          i32x4;

static inline int cdiv(int a, int b) { return (a + b - 1) / b; }

__device__ __forceinline__ float elu1(float x) { return x > 0.f ? x : __expf(x) - 1.f; }

__device__ __forceinline__ bf16x16 join8(bf16x8 lo, bf16x8 hi) {
    return __builtin_shufflevector(lo, hi, 0,1,2,3,4,5,6,7,8,9,10,11,12,13,14,15);
}
__device__ __forceinline__ bf16x16 load_contig16(const __bf16* p) {   // 32 contiguous bytes
    return join8(*(const bf16x8*)p, *(const bf16x8*)(p + 8));
}
__device__ __forceinline__ bf16x16 load_afrag(const __bf16* p) {      // runs at p, p+16 elems
    return join8(*(const bf16x8*)p, *(const bf16x8*)(p + 16));
}

// TDM: 1-D copy of n8*8 bytes from global -> LDS (descriptor per CDNA5 ISA 8.3/8.4)
__device__ __forceinline__ void tdm_copy_1d(const void* gsrc, unsigned ldsOff, unsigned n8) {
    unsigned long long ga = (unsigned long long)(uintptr_t)gsrc;
    u32x4 g0;
    g0[0] = 1u;                                   // count=1, user mode
    g0[1] = ldsOff;                               // lds_addr (bytes)
    g0[2] = (unsigned)ga;                         // global_addr[31:0]
    g0[3] = (unsigned)(ga >> 32) | 0x80000000u;   // global_addr[56:32] | type=2<<30
    i32x8 g1;
    g1[0] = (int)(3u << 16);                      // data_size=3 (8B), no multicast
    g1[1] = (int)((n8 & 0xffffu) << 16);          // tensor_dim0[15:0]
    g1[2] = (int)((n8 >> 16) | (1u << 16));       // tensor_dim0[31:16] | tensor_dim1=1
    g1[3] = (int)((n8 & 0xffffu) << 16);          // tile_dim0 = n8
    g1[4] = 0;                                    // tile_dim1/2 unused
    g1[5] = (int)n8;                              // tensor_dim0_stride
    g1[6] = 0; g1[7] = 0;
    i32x4 z4 = {0, 0, 0, 0};
    i32x8 z8 = {0, 0, 0, 0, 0, 0, 0, 0};
    __builtin_amdgcn_tensor_load_to_lds(g0, g1, z4, z4, z8, 0);
}

// ---------------- trivial kernels ----------------
__global__ void zerof_kernel(float* __restrict__ p, int n) {
    int i = blockIdx.x * blockDim.x + threadIdx.x;
    if (i < n) p[i] = 0.f;
}

__global__ void embed_kernel(const int* __restrict__ x, const int* __restrict__ z,
                             const float* __restrict__ zt, const float* __restrict__ nt,
                             float* __restrict__ h) {
    int idx = blockIdx.x * blockDim.x + threadIdx.x;
    if (idx >= N_NODES * 64) return;
    int n = idx >> 6, c = idx & 63;
    float v = 0.f;
    if (c < 8)       v = nt[x[n] * 8 + c] + zt[z[2 * n] * 8 + c] + zt[z[2 * n + 1] * 8 + c];
    else if (c == 8) v = (float)x[n];
    h[idx] = v;
}

// W2T[(i*Mout+o)*WROW + k] = k<128 ? w2[k][i*Mout+o] : (k==128 ? b2[i*Mout+o] : 0)   (bf16)
__global__ void pack_w2_kernel(const float* __restrict__ w2, const float* __restrict__ b2,
                               __bf16* __restrict__ W2T, int rows /*Min*Mout*/) {
    int idx = blockIdx.x * blockDim.x + threadIdx.x;
    if (idx >= rows * WROW) return;
    int k = idx % WROW, row = idx / WROW;
    float v = 0.f;
    if (k < 128)       v = w2[k * rows + row];
    else if (k == 128) v = b2[row];
    W2T[idx] = (__bf16)v;
}

// Aact[e][c] = c<128 ? relu(edge_attr[e]@w1 + b1)[c] : (c==128 ? 1 : 0)   (bf16)
__global__ void edge_act_kernel(const float* __restrict__ ea, const float* __restrict__ w1,
                                const float* __restrict__ b1, __bf16* __restrict__ Aact) {
    int idx = blockIdx.x * blockDim.x + threadIdx.x;
    if (idx >= N_EDGES * KPAD) return;
    int c = idx % KPAD, e = idx / KPAD;
    float v = 0.f;
    if (c < 128) {
        float t = b1[c];
        #pragma unroll
        for (int j = 0; j < 5; ++j) t += ea[e * 5 + j] * w1[j * 128 + c];
        v = t > 0.f ? t : 0.f;
    } else if (c == 128) v = 1.f;
    Aact[idx] = (__bf16)v;
}

// ---------------- fused NNConv message kernel ----------------
// Block = 8 waves; each wave owns 2 edge tiles (32 edges) -> block covers 256 edges.
// Per input channel i: TDM stages w2T slice [Mout x 336B] into LDS (double buffered);
// every wave computes T = w2T_slice(16o x 160k) @ Aact^T(160k x 16e) with bf16 WMMA,
// then acc[o,e] += h[src_e,i] * T[o,e]; finally atomic scatter-add into agg[dst].
template<int OB>  // Mout = OB*16
__global__ __launch_bounds__(256) void msg_kernel(const __bf16* __restrict__ Aact,
                                                  const __bf16* __restrict__ W2T,
                                                  const int* __restrict__ ei,
                                                  const float* __restrict__ h,
                                                  float* __restrict__ agg, int Min) {
    constexpr int Mout = OB * 16;
    extern __shared__ char smem[];
    __bf16* wbuf0 = (__bf16*)smem;
    __bf16* wbuf1 = wbuf0 + Mout * WROW;
    float*  hT    = (float*)(smem + (size_t)2 * Mout * WROW * 2);   // [HCH][256]

    const int tid  = threadIdx.x;
    const int wave = tid >> 5, lane = tid & 31;
    const int el   = lane & 15, half = lane >> 4;
    const int blockBase = blockIdx.x * 256;
    const bool active = (blockBase + wave * 32) < N_EDGES;          // whole-wave granularity
    const int eb  = active ? (blockBase + wave * 32) : 0;
    const int be0 = eb + el, be1 = eb + 16 + el;
    const int dst0 = ei[N_EDGES + be0];
    const int dst1 = ei[N_EDGES + be1];

    // this thread's gather column for hT staging
    int ge = blockBase + tid; if (ge >= N_EDGES) ge = N_EDGES - 1;
    const int src = ei[ge];

    // B fragments: Aact^T for both edge tiles, K=160 in 5 chunks (register-resident)
    bf16x16 b0[5], b1[5];
    {
        const __bf16* a0 = Aact + (size_t)be0 * KPAD + half * 8;
        const __bf16* a1 = Aact + (size_t)be1 * KPAD + half * 8;
        #pragma unroll
        for (int kc = 0; kc < 5; ++kc) {
            b0[kc] = load_afrag(a0 + kc * 32);
            b1[kc] = load_afrag(a1 + kc * 32);
        }
    }

    // TDM prologue: prefetch slices 0 and 1
    const unsigned n8 = (unsigned)(Mout * WROW * 2 / 8);
    const unsigned lds0 = (unsigned)(uintptr_t)(void*)wbuf0;
    const unsigned lds1 = (unsigned)(uintptr_t)(void*)wbuf1;
    if (wave == 0) {
        tdm_copy_1d(W2T, lds0, n8);
        if (Min > 1) tdm_copy_1d(W2T + (size_t)Mout * WROW, lds1, n8);
    }

    v8f acc[2][OB];
    #pragma unroll
    for (int t = 0; t < 2; ++t)
        #pragma unroll
        for (int ob = 0; ob < OB; ++ob) acc[t][ob] = (v8f){0,0,0,0,0,0,0,0};

    const int lowoff = half * 8;
    for (int i = 0; i < Min; ++i) {
        // refill h[src] chunk (prev iteration's trailing barrier made it safe)
        if ((i & (HCH - 1)) == 0) {
            int pend = Min - i; if (pend > HCH) pend = HCH;
            for (int p = 0; p < pend; ++p) hT[p * 256 + tid] = h[src * 64 + i + p];
        }
        if (wave == 0) {
            if (i < Min - 1) __builtin_amdgcn_s_wait_tensorcnt((short)1);
            else             __builtin_amdgcn_s_wait_tensorcnt((short)0);
        }
        __syncthreads();   // slice i + hT chunk visible to all waves

        const __bf16* wb = (i & 1) ? wbuf1 : wbuf0;
        const float hs0 = hT[(i & (HCH - 1)) * 256 + wave * 32 + el];
        const float hs1 = hT[(i & (HCH - 1)) * 256 + wave * 32 + 16 + el];

        #pragma unroll
        for (int ob = 0; ob < OB; ++ob) {
            const __bf16* wrow = wb + (ob * 16 + el) * WROW + lowoff;
            v8f t0 = (v8f){0,0,0,0,0,0,0,0};
            v8f t1 = (v8f){0,0,0,0,0,0,0,0};
            #pragma unroll
            for (int kc = 0; kc < 5; ++kc) {
                bf16x16 af = load_afrag(wrow + kc * 32);   // 2x ds_load_b128, feeds 2 WMMAs
                t0 = __builtin_amdgcn_wmma_f32_16x16x32_bf16(
                         false, af, false, b0[kc], (short)0, t0, false, false);
                t1 = __builtin_amdgcn_wmma_f32_16x16x32_bf16(
                         false, af, false, b1[kc], (short)0, t1, false, false);
            }
            #pragma unroll
            for (int r = 0; r < 8; ++r) {                  // VALU co-executes with XDL
                acc[0][ob][r] += hs0 * t0[r];
                acc[1][ob][r] += hs1 * t1[r];
            }
        }
        __syncthreads();   // all waves done with slice (i&1) buffer
        if (wave == 0 && i + 2 < Min)
            tdm_copy_1d(W2T + (size_t)(i + 2) * Mout * WROW, (i & 1) ? lds1 : lds0, n8);
    }

    if (active) {
        // D layout: o = ob*16 + r + 8*half, e = lane&15
        const int obase = half * 8;
        #pragma unroll
        for (int ob = 0; ob < OB; ++ob)
            #pragma unroll
            for (int r = 0; r < 8; ++r) {
                unsafeAtomicAdd(&agg[(size_t)dst0 * 64 + ob * 16 + obase + r], acc[0][ob][r]);
                unsafeAtomicAdd(&agg[(size_t)dst1 * 64 + ob * 16 + obase + r], acc[1][ob][r]);
            }
    }
}

// hB[n][o] = elu(agg[n][o] + hA[n][:Min]@root + bias[o]) for o<Mout else 0
__global__ void node_update_kernel(const float* __restrict__ agg, const float* __restrict__ h,
                                   const float* __restrict__ root, const float* __restrict__ bias,
                                   float* __restrict__ out, int Min, int Mout) {
    int idx = blockIdx.x * blockDim.x + threadIdx.x;
    if (idx >= N_NODES * 64) return;
    int n = idx >> 6, o = idx & 63;
    float v = 0.f;
    if (o < Mout) {
        float s = agg[idx] + bias[o];
        for (int i = 0; i < Min; ++i) s += h[n * 64 + i] * root[i * Mout + o];
        v = elu1(s);
    }
    out[idx] = v;
}

// ---------------- pooling + head ----------------
__global__ void pool_scatter_kernel(const float* __restrict__ h, const int* __restrict__ seg,
                                    float* __restrict__ pool, float* __restrict__ cnt, int rows) {
    int idx = blockIdx.x * blockDim.x + threadIdx.x;
    if (idx >= rows * 64) return;
    int n = idx >> 6, c = idx & 63;
    int s = seg[n];
    unsafeAtomicAdd(&pool[s * 64 + c], h[idx]);
    if (c == 0) unsafeAtomicAdd(&cnt[s], 1.f);
}
__global__ void pool_div_kernel(float* __restrict__ pool, const float* __restrict__ cnt, int rows) {
    int idx = blockIdx.x * blockDim.x + threadIdx.x;
    if (idx >= rows * 64) return;
    pool[idx] /= fmaxf(cnt[idx >> 6], 1.f);
}
__global__ void fc_kernel(const float* __restrict__ in, const float* __restrict__ w,
                          const float* __restrict__ b, float* __restrict__ out,
                          int rows, int cin, int cout, int do_elu) {
    int idx = blockIdx.x * blockDim.x + threadIdx.x;
    if (idx >= rows * cout) return;
    int g = idx / cout, o = idx % cout;
    float s = b[o];
    for (int i = 0; i < cin; ++i) s += in[g * cin + i] * w[i * cout + o];
    out[idx] = do_elu ? elu1(s) : s;
}

// ---------------- host orchestration ----------------
extern "C" void kernel_launch(void* const* d_in, const int* in_sizes, int n_in,
                              void* d_out, int out_size, void* d_ws, size_t ws_size,
                              hipStream_t stream) {
    const int* x  = (const int*)d_in[0];
    const int* z  = (const int*)d_in[1];
    const int* ei = (const int*)d_in[2];
    const float* ea = (const float*)d_in[3];
    const int* seg1 = (const int*)d_in[4];
    const int* seg2 = (const int*)d_in[5];
    const int* seg3 = (const int*)d_in[6];
    const float* zt = (const float*)d_in[7];
    const float* nt = (const float*)d_in[8];
    const float* fc1w = (const float*)d_in[39]; const float* fc1b = (const float*)d_in[40];
    const float* fc2w = (const float*)d_in[41]; const float* fc2b = (const float*)d_in[42];
    const float* fc3w = (const float*)d_in[43]; const float* fc3b = (const float*)d_in[44];
    float* out = (float*)d_out;

    char* ws = (char*)d_ws;
    size_t off = 0;
    auto carve = [&](size_t bytes) { void* p = ws + off; off = (off + bytes + 255) & ~(size_t)255; return p; };
    float*  hA   = (float*) carve((size_t)N_NODES * 64 * 4);
    float*  hB   = (float*) carve((size_t)N_NODES * 64 * 4);
    float*  agg  = (float*) carve((size_t)N_NODES * 64 * 4);
    __bf16* Aact = (__bf16*)carve((size_t)N_EDGES * KPAD * 2);
    __bf16* W2T  = (__bf16*)carve((size_t)64 * 64 * WROW * 2);
    float*  p1   = (float*) carve((size_t)N_SG2 * 64 * 4);
    float*  c1   = (float*) carve((size_t)N_SG2 * 4);
    float*  p2   = (float*) carve((size_t)N_SG * 64 * 4);
    float*  c2   = (float*) carve((size_t)N_SG * 4);
    float*  p3   = (float*) carve((size_t)N_GRAPHS * 64 * 4);
    float*  c3   = (float*) carve((size_t)N_GRAPHS * 4);
    float*  t1   = (float*) carve((size_t)N_GRAPHS * 32 * 4);
    float*  t2   = (float*) carve((size_t)N_GRAPHS * 16 * 4);
    (void)ws_size; (void)n_in; (void)out_size;

    const int B = 256;
    embed_kernel<<<cdiv(N_NODES * 64, B), B, 0, stream>>>(x, z, zt, nt, hA);

    static const int MinL[5]  = {9, 32, 64, 64, 64};
    static const int MoutL[5] = {32, 64, 64, 64, 64};
    const float* cur = hA;
    float* nxt = hB;

    for (int l = 0; l < 5; ++l) {
        int base = 9 + 6 * l;
        const float *w1, *b1, *w2, *b2, *root, *bias;
        if (in_sizes[base] == 640) {  // insertion order: w1,b1,w2,b2,root,bias
            w1   = (const float*)d_in[base + 0]; b1  = (const float*)d_in[base + 1];
            w2   = (const float*)d_in[base + 2]; b2  = (const float*)d_in[base + 3];
            root = (const float*)d_in[base + 4]; bias= (const float*)d_in[base + 5];
        } else {                       // sorted-key order: b1,b2,bias,root,w1,w2
            b1   = (const float*)d_in[base + 0]; b2  = (const float*)d_in[base + 1];
            bias = (const float*)d_in[base + 2]; root= (const float*)d_in[base + 3];
            w1   = (const float*)d_in[base + 4]; w2  = (const float*)d_in[base + 5];
        }
        int Min = MinL[l], Mout = MoutL[l], rows = Min * Mout;

        pack_w2_kernel<<<cdiv(rows * WROW, B), B, 0, stream>>>(w2, b2, W2T, rows);
        edge_act_kernel<<<cdiv(N_EDGES * KPAD, B), B, 0, stream>>>(ea, w1, b1, Aact);
        zerof_kernel<<<cdiv(N_NODES * 64, B), B, 0, stream>>>(agg, N_NODES * 64);

        int nblk = cdiv(N_EDGES, 256);
        size_t smem = (size_t)2 * Mout * WROW * 2 + (size_t)HCH * 256 * 4;
        if (Mout == 32)
            msg_kernel<2><<<nblk, 256, smem, stream>>>(Aact, W2T, ei, cur, agg, Min);
        else
            msg_kernel<4><<<nblk, 256, smem, stream>>>(Aact, W2T, ei, cur, agg, Min);

        node_update_kernel<<<cdiv(N_NODES * 64, B), B, 0, stream>>>(agg, cur, root, bias, nxt, Min, Mout);
        const float* tmp = cur; cur = nxt; nxt = (float*)tmp;
    }

    zerof_kernel<<<cdiv(N_SG2 * 64, B), B, 0, stream>>>(p1, N_SG2 * 64);
    zerof_kernel<<<cdiv(N_SG2, B), B, 0, stream>>>(c1, N_SG2);
    pool_scatter_kernel<<<cdiv(N_NODES * 64, B), B, 0, stream>>>(cur, seg1, p1, c1, N_NODES);
    pool_div_kernel<<<cdiv(N_SG2 * 64, B), B, 0, stream>>>(p1, c1, N_SG2);

    zerof_kernel<<<cdiv(N_SG * 64, B), B, 0, stream>>>(p2, N_SG * 64);
    zerof_kernel<<<cdiv(N_SG, B), B, 0, stream>>>(c2, N_SG);
    pool_scatter_kernel<<<cdiv(N_SG2 * 64, B), B, 0, stream>>>(p1, seg2, p2, c2, N_SG2);
    pool_div_kernel<<<cdiv(N_SG * 64, B), B, 0, stream>>>(p2, c2, N_SG);

    zerof_kernel<<<cdiv(N_GRAPHS * 64, B), B, 0, stream>>>(p3, N_GRAPHS * 64);
    zerof_kernel<<<cdiv(N_GRAPHS, B), B, 0, stream>>>(c3, N_GRAPHS);
    pool_scatter_kernel<<<cdiv(N_SG * 64, B), B, 0, stream>>>(p2, seg3, p3, c3, N_SG);
    pool_div_kernel<<<cdiv(N_GRAPHS * 64, B), B, 0, stream>>>(p3, c3, N_GRAPHS);

    fc_kernel<<<cdiv(N_GRAPHS * 32, B), B, 0, stream>>>(p3, fc1w, fc1b, t1, N_GRAPHS, 64, 32, 1);
    fc_kernel<<<cdiv(N_GRAPHS * 16, B), B, 0, stream>>>(t1, fc2w, fc2b, t2, N_GRAPHS, 32, 16, 1);
    fc_kernel<<<cdiv(N_GRAPHS, B), B, 0, stream>>>(t2, fc3w, fc3b, out, N_GRAPHS, 16, 1, 0);
}
